// SAP2_59347858096135
// MI455X (gfx1250) — matrix operand order repaired and checked by reading, not compiled, not run on hardware
//
#include <hip/hip_runtime.h>
#include <hip/hip_bf16.h>

// ---------------------------------------------------------------------------
// LSTM (B=256,T=512,D=128,H=256) + per-step MLP head, persistent-WGP design.
// 16 persistent workgroups (one 16-row WMMA M-tile each), zero inter-WG sync.
// Recurrent state in LDS/registers; weights pre-packed to WMMA fragment layout
// and streamed from L2 via explicit global_load_b128 (immediate k-tile
// offsets, 2-deep software pipeline, SE/WGP-scope prefetch). x tiles are
// prefetched into a single LDS buffer with the Tensor Data Mover.
// ---------------------------------------------------------------------------

typedef __bf16 bf16_t;
typedef __attribute__((ext_vector_type(16))) __bf16 v16bf;
typedef __attribute__((ext_vector_type(8)))  float  v8f;
typedef unsigned int  u32x4 __attribute__((ext_vector_type(4)));
typedef int           i32x8 __attribute__((ext_vector_type(8)));
typedef int           i32x4 __attribute__((ext_vector_type(4)));

#define B_    256
#define T_    512
#define D_    128
#define H_    256
#define G_    1024            // 4H
#define NTK_G 12              // (D+H)/32 k-tiles for gates GEMM
#define NTN_G 64              // 4H/16 n-tiles
#define NTK_M 8               // H/32
#define NTN_M 16              // H/16
#define TILE_ELEMS 512        // 32 lanes * 16 bf16 per fragment-pair tile
#define TILE_BYTES 1024       // per k-tile stride in the packed arrays

#ifndef __has_builtin
#define __has_builtin(x) 0
#endif
#if __has_builtin(__builtin_amdgcn_tensor_load_to_lds) && \
    __has_builtin(__builtin_amdgcn_s_wait_tensorcnt)
#define USE_TDM 1
#else
#define USE_TDM 0
#endif

// A-fragment packing: lane<16 holds row m = lane, K = {0..7,16..23};
// lane>=16 holds row m = lane-16, K = {8..15,24..31}.
__device__ __forceinline__ int apack_idx(int m, int q) {
  int lane = m + (((q >> 3) & 1) << 4);
  int slot = (q & 7) + ((q >> 4) << 3);
  return lane * 16 + slot;
}

// ---- explicit weight-fragment loads: single base VGPR pair, k-tile offset
//      folded into the instruction's signed 24-bit immediate (no VALU adds,
//      no register churn -> no WMMA->VALU hazard NOPs).
__device__ __forceinline__ void frag_issue_off(unsigned long long base, int byteoff,
                                               i32x4& lo, i32x4& hi) {
  asm volatile("global_load_b128 %0, %2, off offset:%3\n\t"
               "global_load_b128 %1, %2, off offset:%4"
               : "=v"(lo), "=v"(hi)
               : "v"(base), "i"(byteoff), "i"(byteoff + 16));
}
// wait until LOADcnt <= N; "touch" fragments so consumers depend on the wait
template <int N>
__device__ __forceinline__ void wait_touch4(i32x4& a, i32x4& b, i32x4& c, i32x4& d) {
  asm volatile("s_wait_loadcnt %4"
               : "+v"(a), "+v"(b), "+v"(c), "+v"(d)
               : "i"(N));
}
template <int N>
__device__ __forceinline__ void wait_touch2(i32x4& a, i32x4& b) {
  asm volatile("s_wait_loadcnt %2" : "+v"(a), "+v"(b) : "i"(N));
}
__device__ __forceinline__ v16bf as_frag(const i32x4& lo, const i32x4& hi) {
  union { i32x4 i[2]; v16bf v; } u;
  u.i[0] = lo; u.i[1] = hi;
  return u.v;
}
__device__ __forceinline__ v8f wmma_bf16(v16bf a, v16bf b, v8f c) {
  return __builtin_amdgcn_wmma_f32_16x16x32_bf16(false, a, false, b,
                                                 (short)0, c, false, false);
}

// ---------------- weight packing: B-fragment layout --------------------------
// layout: [ntile][ktile][lane(32)][slot(16)], lane -> (n = lane&15, khalf),
// slot s -> k_local = (s&7) + 8*khalf + 16*(s>>3)  (contiguous 32B per lane per fragment).
__global__ void pack_weights(const float* __restrict__ Wx, const float* __restrict__ Wh,
                             const float* __restrict__ W1, const float* __restrict__ W2,
                             bf16_t* __restrict__ Wxhpk, bf16_t* __restrict__ W1pk,
                             bf16_t* __restrict__ W2pk) {
  int idx = blockIdx.x * 256 + threadIdx.x;
  const int NWXH = NTN_G * NTK_G * TILE_ELEMS;   // 393216
  const int NW1  = NTN_M * NTK_M * TILE_ELEMS;   // 65536
  if (idx < NWXH) {
    int slot = idx & 15, lane = (idx >> 4) & 31, tile = idx >> 9;
    int tk = tile % NTK_G, tn = tile / NTK_G;
    int n = tn * 16 + (lane & 15);
    int k = tk * 32 + (slot & 7) + (lane >> 4) * 8 + (slot >> 3) * 16;   // concat K: [x(0..127); h(128..383)]
    float v = (k < D_) ? Wx[k * G_ + n] : Wh[(k - D_) * G_ + n];
    Wxhpk[idx] = (bf16_t)v;
  } else {
    int j = idx - NWXH;
    const float* W = (j < NW1) ? W1 : W2;
    bf16_t* dst    = (j < NW1) ? W1pk : W2pk;
    int jj = (j < NW1) ? j : j - NW1;
    int slot = jj & 15, lane = (jj >> 4) & 31, tile = jj >> 9;
    int tk = tile % NTK_M, tn = tile / NTK_M;
    int n = tn * 16 + (lane & 15);
    int k = tk * 32 + (slot & 7) + (lane >> 4) * 8 + (slot >> 3) * 16;
    dst[jj] = (bf16_t)W[k * H_ + n];
  }
}

// ---------------- x packing: per-(t, row-tile) A-fragment layout -------------
// xpk layout: [t][tm][tk(4)][lane(32)][slot(16)]  (2048 bf16 = 4KB per (t,tm))
__global__ void pack_x(const float* __restrict__ x, bf16_t* __restrict__ xpk) {
  long idx = (long)blockIdx.x * 256 + threadIdx.x;   // 0 .. 16777215
  int slot = (int)(idx & 15), lane = (int)((idx >> 4) & 31);
  long tile = idx >> 9;
  int tk = (int)(tile & 3);
  int tm = (int)((tile >> 2) & 15);
  int t  = (int)(tile >> 6);
  int m    = lane & 15;
  int half = lane >> 4;
  int q = (slot & 7) + 8 * half + 16 * (slot >> 3);
  int b = tm * 16 + m;
  int d = tk * 32 + q;
  xpk[idx] = (bf16_t)x[((long)b * T_ + t) * D_ + d];
}

// ---------------- persistent LSTM + MLP kernel -------------------------------
__global__ __launch_bounds__(1024) void lstm_persistent(
    const bf16_t* __restrict__ xpk, const bf16_t* __restrict__ Wxhpk,
    const bf16_t* __restrict__ W1pk, const bf16_t* __restrict__ W2pk,
    const float* __restrict__ bg, const float* __restrict__ b1,
    const float* __restrict__ b2, const float* __restrict__ W3,
    const float* __restrict__ b3, float* __restrict__ out) {
  __shared__ bf16_t s_gates[16 * G_];   // 32 KB  pre-activations (16 x 1024)
  __shared__ bf16_t s_hpk[8 * 512];     //  8 KB  packed h (A layout)
  __shared__ bf16_t s_xt[4 * 512];      //  4 KB  packed x_t (A layout)
  __shared__ bf16_t s_y1pk[8 * 512];    //  8 KB  packed y1
  __shared__ bf16_t s_y2[16 * H_];      //  8 KB  y2 plain [16][256]

  const int tid  = threadIdx.x;
  const int lane = tid & 31;
  const int wave = tid >> 5;            // 32 wave32 waves
  const int wg   = blockIdx.x;          // batch rows wg*16 .. wg*16+15

  // --- init: zero packed h, load x tile for t=0 ---
  {
    uint32_t* hz = (uint32_t*)s_hpk;    // 2048 dwords
    hz[tid * 2 + 0] = 0u;
    hz[tid * 2 + 1] = 0u;
    if (tid < 512) {
      const uint2* src = (const uint2*)(xpk + (size_t)wg * 2048);
      ((uint2*)s_xt)[tid] = src[tid];
    }
  }
  float creg[4] = {0.f, 0.f, 0.f, 0.f};
  const float bias3 = b3[0];
  __syncthreads();

  const bf16_t* bptr0 = Wxhpk + ((size_t)(wave * 2 + 0) * NTK_G) * TILE_ELEMS + lane * 16;
  const bf16_t* bptr1 = Wxhpk + ((size_t)(wave * 2 + 1) * NTK_G) * TILE_ELEMS + lane * 16;
  const bf16_t* w1ptr = W1pk + ((size_t)wave * NTK_M) * TILE_ELEMS + lane * 16;
  const bf16_t* w2ptr = W2pk + ((size_t)wave * NTK_M) * TILE_ELEMS + lane * 16;
  const unsigned long long b0a = (unsigned long long)(uintptr_t)bptr0;
  const unsigned long long b1a = (unsigned long long)(uintptr_t)bptr1;
  const unsigned long long w1a = (unsigned long long)(uintptr_t)w1ptr;
  const unsigned long long w2a = (unsigned long long)(uintptr_t)w2ptr;

  for (int t = 0; t < T_; ++t) {
    // ---- Phase A: gates = [x_t ; h] @ [Wx;Wh] + b  (each wave: 2 N-tiles)
    {
      i32x4 fb[2][2][2];                 // [stage][ntile][lo/hi]
      frag_issue_off(b0a, 0, fb[0][0][0], fb[0][0][1]);
      frag_issue_off(b1a, 0, fb[0][1][0], fb[0][1][1]);
      v8f acc0 = {0.f,0.f,0.f,0.f,0.f,0.f,0.f,0.f};
      v8f acc1 = {0.f,0.f,0.f,0.f,0.f,0.f,0.f,0.f};
#pragma unroll
      for (int kt = 0; kt < NTK_G; ++kt) {
        const int cur = kt & 1, nxt = cur ^ 1;
        if (kt + 2 < NTK_G) {            // prefetch, 2 tiles ahead
          __builtin_prefetch(bptr0 + (kt + 2) * TILE_ELEMS, 0, 3);
          __builtin_prefetch(bptr1 + (kt + 2) * TILE_ELEMS, 0, 3);
        }
        if (kt + 1 < NTK_G) {
          frag_issue_off(b0a, (kt + 1) * TILE_BYTES, fb[nxt][0][0], fb[nxt][0][1]);
          frag_issue_off(b1a, (kt + 1) * TILE_BYTES, fb[nxt][1][0], fb[nxt][1][1]);
          wait_touch4<4>(fb[cur][0][0], fb[cur][0][1], fb[cur][1][0], fb[cur][1][1]);
        } else {
          wait_touch4<0>(fb[cur][0][0], fb[cur][0][1], fb[cur][1][0], fb[cur][1][1]);
        }
        const bf16_t* asrc = (kt < 4) ? &s_xt[kt * 512 + lane * 16]
                                      : &s_hpk[(kt - 4) * 512 + lane * 16];
        v16bf a  = *(const v16bf*)asrc;
        acc0 = wmma_bf16(a, as_frag(fb[cur][0][0], fb[cur][0][1]), acc0);
        acc1 = wmma_bf16(a, as_frag(fb[cur][1][0], fb[cur][1][1]), acc1);
      }
      const int nl = lane & 15;
      const int mb = 8 * (lane >> 4);
      {
        const int ng = (wave * 2 + 0) * 16 + nl;
        const float bi = bg[ng];
#pragma unroll
        for (int r = 0; r < 8; ++r)
          s_gates[(mb + r) * G_ + ng] = (bf16_t)(acc0[r] + bi);
      }
      {
        const int ng = (wave * 2 + 1) * 16 + nl;
        const float bi = bg[ng];
#pragma unroll
        for (int r = 0; r < 8; ++r)
          s_gates[(mb + r) * G_ + ng] = (bf16_t)(acc1[r] + bi);
      }
    }
    __syncthreads();

    // ---- Phase B: elementwise LSTM cell; writes packed h into LDS
#pragma unroll
    for (int p = 0; p < 4; ++p) {
      const int idx = tid + 1024 * p;       // fixed ownership -> c stays in regs
      const int j = idx & 255, b = idx >> 8;
      float i = (float)s_gates[b * G_ + j];
      float f = (float)s_gates[b * G_ + 256 + j];
      float g = (float)s_gates[b * G_ + 512 + j];
      float o = (float)s_gates[b * G_ + 768 + j];
      i = 1.f / (1.f + __expf(-i));
      f = 1.f / (1.f + __expf(-f));
      o = 1.f / (1.f + __expf(-o));
      float cn = f * creg[p] + i * tanhf(g);
      creg[p] = cn;
      float h = o * tanhf(cn);
      s_hpk[(j >> 5) * 512 + apack_idx(b, j & 31)] = (bf16_t)h;
    }
    __syncthreads();

    // ---- Phase C: y1 = tanh(h @ W1 + b1) -> packed (waves 0..15);
    //      wave 16 prefetches next x tile into s_xt (TDM when available)
    if (wave < 16) {
      i32x4 fc[2][2];
      frag_issue_off(w1a, 0, fc[0][0], fc[0][1]);
      v8f acc = {0.f,0.f,0.f,0.f,0.f,0.f,0.f,0.f};
#pragma unroll
      for (int kt = 0; kt < NTK_M; ++kt) {
        const int cur = kt & 1, nxt = cur ^ 1;
        if (kt + 2 < NTK_M) __builtin_prefetch(w1ptr + (kt + 2) * TILE_ELEMS, 0, 3);
        if (kt + 1 < NTK_M) {
          frag_issue_off(w1a, (kt + 1) * TILE_BYTES, fc[nxt][0], fc[nxt][1]);
          wait_touch2<2>(fc[cur][0], fc[cur][1]);
        } else {
          wait_touch2<0>(fc[cur][0], fc[cur][1]);
        }
        v16bf a = *(const v16bf*)&s_hpk[kt * 512 + lane * 16];
        acc = wmma_bf16(a, as_frag(fc[cur][0], fc[cur][1]), acc);
      }
      const int ng = wave * 16 + (lane & 15);
      const float bi = b1[ng];
      const int tk = ng >> 5, q = ng & 31;
      const int ps = (q & 7) + ((q >> 4) << 3);
      const int lh = ((q >> 3) & 1) << 4;
      const int mb = 8 * (lane >> 4);
#pragma unroll
      for (int r = 0; r < 8; ++r)
        s_y1pk[tk * 512 + (mb + r + lh) * 16 + ps] = (bf16_t)tanhf(acc[r] + bi);
    } else {
      const int tn = t + 1;
      if (tn < T_) {
#if USE_TDM
        if (wave == 16) {
          // Tensor Data Mover: 1-D tile, 512 x 8B = 4KB  global -> s_xt
          unsigned long long ga =
              (unsigned long long)(uintptr_t)(xpk + (size_t)(tn * 16 + wg) * 2048);
          unsigned lds = (unsigned)(uintptr_t)(&s_xt[0]);
          u32x4 g0;
          g0[0] = 1u;                                    // count=1, load descriptor
          g0[1] = lds;                                   // lds_addr
          g0[2] = (unsigned)(ga & 0xFFFFFFFFu);          // global_addr[31:0]
          g0[3] = (unsigned)((ga >> 32) & 0x01FFFFFFu)   // global_addr[56:32]
                  | (2u << 30);                          // type = 2 ("image")
          i32x8 g1;
          g1[0] = (3 << 16);                             // data_size = 8B, mask=0
          g1[1] = (int)(512u << 16);                     // tensor_dim0[15:0] -> bits 63:48
          g1[2] = 0;                                     // tensor_dim0 hi, tensor_dim1 lo
          g1[3] = (int)(512u << 16);                     // tile_dim0 = 512 (bits 127:112)
          g1[4] = 0;                                     // tile_dim1/2 unused (1-D)
          g1[5] = 512;                                   // tensor_dim0_stride
          g1[6] = 0;
          g1[7] = 0;
          i32x4 gz4 = {0, 0, 0, 0};
          i32x8 gz8 = {0, 0, 0, 0, 0, 0, 0, 0};
          // 6-arg form (clang-23 / therock-10.0 headers)
          __builtin_amdgcn_tensor_load_to_lds(g0, g1, gz4, gz4, gz8, 0);
        }
#else
        const int tid2 = tid - 512;
        const uint2* src = (const uint2*)(xpk + (size_t)(tn * 16 + wg) * 2048);
        ((uint2*)s_xt)[tid2] = src[tid2];
#endif
      }
    }
    __syncthreads();

    // ---- Phase D: y2 = tanh(y1 @ W2 + b2) -> plain [16][256] bf16
    if (wave < 16) {
      i32x4 fd[2][2];
      frag_issue_off(w2a, 0, fd[0][0], fd[0][1]);
      v8f acc = {0.f,0.f,0.f,0.f,0.f,0.f,0.f,0.f};
#pragma unroll
      for (int kt = 0; kt < NTK_M; ++kt) {
        const int cur = kt & 1, nxt = cur ^ 1;
        if (kt + 2 < NTK_M) __builtin_prefetch(w2ptr + (kt + 2) * TILE_ELEMS, 0, 3);
        if (kt + 1 < NTK_M) {
          frag_issue_off(w2a, (kt + 1) * TILE_BYTES, fd[nxt][0], fd[nxt][1]);
          wait_touch2<2>(fd[cur][0], fd[cur][1]);
        } else {
          wait_touch2<0>(fd[cur][0], fd[cur][1]);
        }
        v16bf a = *(const v16bf*)&s_y1pk[kt * 512 + lane * 16];
        acc = wmma_bf16(a, as_frag(fd[cur][0], fd[cur][1]), acc);
      }
      const int ng = wave * 16 + (lane & 15);
      const float bi = b2[ng];
      const int mb = 8 * (lane >> 4);
#pragma unroll
      for (int r = 0; r < 8; ++r)
        s_y2[(mb + r) * H_ + ng] = (bf16_t)tanhf(acc[r] + bi);
    }
    __syncthreads();

    // ---- Phase E: out[b][t] = y2[b,:].W3 + b3  (waves 0..15, one row each)
    if (wave < 16) {
      float sum = 0.f;
#pragma unroll
      for (int k = 0; k < 8; ++k) {
        const int col = lane + 32 * k;
        sum += (float)s_y2[wave * H_ + col] * W3[col];
      }
#pragma unroll
      for (int off = 16; off > 0; off >>= 1)
        sum += __shfl_xor(sum, off, 32);
      if (lane == 0)
        out[(size_t)(wg * 16 + wave) * T_ + t] = sum + bias3;
    }
#if USE_TDM
    __builtin_amdgcn_s_wait_tensorcnt(0);   // no-op for waves with TENSORcnt==0
#endif
    __syncthreads();
  }
}

// ---------------------------------------------------------------------------
extern "C" void kernel_launch(void* const* d_in, const int* in_sizes, int n_in,
                              void* d_out, int out_size, void* d_ws, size_t ws_size,
                              hipStream_t stream) {
  const float* x  = (const float*)d_in[0];
  const float* Wx = (const float*)d_in[1];
  const float* Wh = (const float*)d_in[2];
  const float* bg = (const float*)d_in[3];
  const float* W1 = (const float*)d_in[4];
  const float* b1 = (const float*)d_in[5];
  const float* W2 = (const float*)d_in[6];
  const float* b2 = (const float*)d_in[7];
  const float* W3 = (const float*)d_in[8];
  const float* b3 = (const float*)d_in[9];
  float* out = (float*)d_out;

  // workspace layout (bytes): Wxhpk 768K | W1pk 128K | W2pk 128K | xpk 32M
  char* ws = (char*)d_ws;
  bf16_t* Wxhpk = (bf16_t*)(ws);
  bf16_t* W1pk  = (bf16_t*)(ws + 786432);
  bf16_t* W2pk  = (bf16_t*)(ws + 917504);
  bf16_t* xpk   = (bf16_t*)(ws + 1048576);

  pack_weights<<<2048, 256, 0, stream>>>(Wx, Wh, W1, W2, Wxhpk, W1pk, W2pk);
  pack_x<<<65536, 256, 0, stream>>>(x, xpk);
  lstm_persistent<<<16, 1024, 0, stream>>>(xpk, Wxhpk, W1pk, W2pk,
                                           bg, b1, b2, W3, b3, out);
}
